// GPT_27333171872554
// MI455X (gfx1250) — matrix operand order repaired and checked by reading, not compiled, not run on hardware
//
#include <hip/hip_runtime.h>
#include <cstdint>

#define DEVINL __device__ __forceinline__

typedef _Float16 half_t;
typedef __attribute__((ext_vector_type(16))) _Float16 v16h;
typedef __attribute__((ext_vector_type(8)))  _Float16 v8h;
typedef __attribute__((ext_vector_type(8)))  float    v8f;
typedef __attribute__((ext_vector_type(4)))  unsigned int v4u;
typedef __attribute__((ext_vector_type(8)))  int v8i;
typedef __attribute__((ext_vector_type(4)))  int v4i;

#if __has_builtin(__builtin_amdgcn_tensor_load_to_lds)
#define HAVE_TDM 1
#else
#define HAVE_TDM 0
#endif

constexpr int LNUM  = 12;
constexpr int HH    = 8;
constexpr int C     = 256;
constexpr int DHD   = 32;
constexpr int TT    = 827;
constexpr int BB    = 4;
constexpr int BT    = BB * TT;       // 3308
constexpr int VOCAB = 16384;
constexpr int FF    = 1024;
constexpr int TP    = 832;           // padded key length (multiple of 32)

// ---------------------------------------------------------------------------
// helpers
// ---------------------------------------------------------------------------
DEVINL float geluf(float x) {
  return 0.5f * x * (1.0f + erff(x * 0.70710678118654752f));
}

DEVINL float exp2f_fast(float x) { return __builtin_amdgcn_exp2f(x); }  // v_exp_f32

DEVINL v8f wmma16(v16h a, v16h b, v8f c) {
  return __builtin_amdgcn_wmma_f32_16x16x32_f16(
      false, a, false, b, (short)0, c, false, false);
}

// A-matrix tile (16x32) direct from global, row stride `ld`.
DEVINL v16h load_A(const half_t* __restrict__ X, int ld, int row0, int k0, int rMax) {
  int l = threadIdx.x & 31;
  int r = row0 + (l & 15);
  if (r > rMax) r = rMax;
  const half_t* p = X + (size_t)r * ld + k0 + ((l >> 4) << 4);
  v8h lo = *(const v8h*)p;
  v8h hi = *(const v8h*)(p + 8);
  v16h a;
#pragma unroll
  for (int i = 0; i < 8; i++) { a[i] = lo[i]; a[8 + i] = hi[i]; }
  return a;
}

// B-matrix tile (32x16) direct from global: B(k,n) = W[n][k].
DEVINL v16h load_B(const half_t* __restrict__ W, int ld, int n0, int k0, int nMax) {
  int l = threadIdx.x & 31;
  int n = n0 + (l & 15);
  if (n > nMax) n = nMax;
  return *(const v16h*)(W + (size_t)n * ld + k0 + ((l >> 4) << 4));
}

// Read a staged row-major LDS tile (rows x 32 halfs) in the WMMA A-layout.
DEVINL v16h lds_read_A(const half_t* buf, int rowBase) {
  int l = threadIdx.x & 31;
  const half_t* p = buf + (size_t)(rowBase + (l & 15)) * 32 + ((l >> 4) << 4);
  v8h lo = *(const v8h*)p;
  v8h hi = *(const v8h*)(p + 8);
  v16h a;
#pragma unroll
  for (int i = 0; i < 8; i++) { a[i] = lo[i]; a[8 + i] = hi[i]; }
  return a;
}

// Read a staged row-major LDS tile in the WMMA B-layout (row n holds K).
DEVINL v16h lds_read_B(const half_t* buf, int n0) {
  int l = threadIdx.x & 31;
  return *(const v16h*)(buf + (size_t)(n0 + (l & 15)) * 32 + ((l >> 4) << 4));
}

// XOR-shuffle reduction within 16-lane groups via ds_swizzle (imm pattern).
#define SWZ_XOR(v, m) \
  __int_as_float(__builtin_amdgcn_ds_swizzle(__float_as_int(v), ((m) << 10) | 0x1f))
DEVINL float redmax16(float v) {
  v = fmaxf(v, SWZ_XOR(v, 1));
  v = fmaxf(v, SWZ_XOR(v, 2));
  v = fmaxf(v, SWZ_XOR(v, 4));
  v = fmaxf(v, SWZ_XOR(v, 8));
  return v;
}
DEVINL float redsum16(float v) {
  v += SWZ_XOR(v, 1);
  v += SWZ_XOR(v, 2);
  v += SWZ_XOR(v, 4);
  v += SWZ_XOR(v, 8);
  return v;
}

// Block-cooperative A staging: 64 rows x 32 halfs (4KB), one async b128 per wave.
DEVINL void stage_A_block(const half_t* __restrict__ X, int ld, int row0blk, int k0,
                          int rMax, half_t* dstBase) {
  int l = threadIdx.x & 31, w = threadIdx.x >> 5;
  int rl = w * 8 + (l >> 2);
  int r = row0blk + rl;
  if (r > rMax) r = rMax;
  int cb = (l & 3) * 8;  // halfs (16B)
  uint64_t g = (uint64_t)(uintptr_t)(X + (size_t)r * ld + k0 + cb);
  uint32_t d = (uint32_t)(uintptr_t)(dstBase + rl * 32 + cb);
  asm volatile("global_load_async_to_lds_b128 %0, %1, off" ::"v"(d), "v"(g) : "memory");
}

// Fallback B staging (128 rows x 32 halfs, 2 async ops per wave).
DEVINL void stage_B_block_async(const half_t* __restrict__ W, int ld, int n0, int k0,
                                half_t* dstBase) {
  int l = threadIdx.x & 31, w = threadIdx.x >> 5;
  int cb = (l & 3) * 8;
#pragma unroll
  for (int s = 0; s < 2; s++) {
    int rl = w * 16 + s * 8 + (l >> 2);
    uint64_t g = (uint64_t)(uintptr_t)(W + (size_t)(n0 + rl) * ld + k0 + cb);
    uint32_t d = (uint32_t)(uintptr_t)(dstBase + rl * 32 + cb);
    asm volatile("global_load_async_to_lds_b128 %0, %1, off" ::"v"(d), "v"(g) : "memory");
  }
}

#if HAVE_TDM
// TDM: stage a 128(N-rows) x 32(K) f16 tile into LDS. One instruction, wave 0 only.
DEVINL void stage_B_tdm(const half_t* W, int N, int K, int n0, int k0, half_t* dstBase) {
  uint64_t ga = (uint64_t)(uintptr_t)(W + (size_t)n0 * K + k0);
  uint32_t lofs = (uint32_t)(uintptr_t)dstBase;
  v4u g0 = { 0x1u,                                   // count=1, user descriptor
             lofs,                                   // lds_addr
             (uint32_t)ga,                           // global_addr[31:0]
             (uint32_t)(ga >> 32) | (2u << 30) };    // global_addr[56:32], type=2
  v8i g1 = { (int)0x10000,                                          // data_size=2B
             (int)(((unsigned)K & 0xFFFFu) << 16),                  // tensor_dim0 lo
             (int)(((unsigned)K >> 16) | (((unsigned)N & 0xFFFFu) << 16)),  // dim0 hi|dim1 lo
             (int)(((unsigned)N >> 16) | (32u << 16)),              // dim1 hi | tile_dim0=32
             (int)128,                                              // tile_dim1=128
             (int)K,                                                // tensor_dim0_stride lo
             0, 0 };
  v4i z4 = {};
  v8i z8 = {};
  __builtin_amdgcn_tensor_load_to_lds(g0, g1, z4, z4, z8, 0);  // 6-arg (clang-23)
}
#endif

// ---------------------------------------------------------------------------
// WMMA GEMM: Y = act(X @ W^T + bias) [+ residual]
// X: (M,K) f16 row-major, W: (N,K) f16 row-major, N multiple of 128.
// Block tile 64(M) x 128(N); 8 waves in 2x4; 32x32 per wave (4 WMMA/k-step).
// A staged via async-VMEM DMA, B via Tensor Data Mover; double buffered.
// ---------------------------------------------------------------------------
template <int ACT, int OUT16, int RES>
__global__ __launch_bounds__(256) void gemm_kernel(
    const half_t* __restrict__ X, const half_t* __restrict__ W,
    const float* __restrict__ bias, float* __restrict__ Yf,
    half_t* __restrict__ Yh, const float* __restrict__ resid,
    int M, int N, int K) {
  __shared__ __align__(128) half_t Ab[2][64 * 32];    //  8 KB
  __shared__ __align__(128) half_t Bb[2][128 * 32];   // 16 KB
  const int lane = threadIdx.x & 31;
  const int wave = threadIdx.x >> 5;
  const int mi = wave & 1;        // 0..1 : 32-row sub-tile
  const int ni = wave >> 1;       // 0..3 : 32-col sub-tile
  const int rowBlk = blockIdx.y * 64;
  const int colBlk = blockIdx.x * 128;
  const int rMax = M - 1;

  // prologue: stage k0 = 0
  stage_A_block(X, K, rowBlk, 0, rMax, Ab[0]);
#if HAVE_TDM
  if (wave == 0) stage_B_tdm(W, N, K, colBlk, 0, Bb[0]);
#else
  stage_B_block_async(W, K, colBlk, 0, Bb[0]);
#endif
  asm volatile("s_wait_asynccnt 0" ::: "memory");
#if HAVE_TDM
  if (wave == 0) __builtin_amdgcn_s_wait_tensorcnt(0);
#endif
  __syncthreads();

  v8f acc[2][2] = {};
  int par = 0;
  for (int k0 = 0; k0 < K; k0 += 32) {
    const bool more = (k0 + 32 < K);
    if (more) {
      stage_A_block(X, K, rowBlk, k0 + 32, rMax, Ab[par ^ 1]);
#if HAVE_TDM
      if (wave == 0) stage_B_tdm(W, N, K, colBlk, k0 + 32, Bb[par ^ 1]);
#else
      stage_B_block_async(W, K, colBlk, k0 + 32, Bb[par ^ 1]);
#endif
    }
    v16h a0 = lds_read_A(Ab[par], mi * 32);
    v16h a1 = lds_read_A(Ab[par], mi * 32 + 16);
    v16h b0 = lds_read_B(Bb[par], ni * 32);
    v16h b1 = lds_read_B(Bb[par], ni * 32 + 16);
    acc[0][0] = wmma16(a0, b0, acc[0][0]);
    acc[0][1] = wmma16(a0, b1, acc[0][1]);
    acc[1][0] = wmma16(a1, b0, acc[1][0]);
    acc[1][1] = wmma16(a1, b1, acc[1][1]);
    if (more) {
      asm volatile("s_wait_asynccnt 0" ::: "memory");
#if HAVE_TDM
      if (wave == 0) __builtin_amdgcn_s_wait_tensorcnt(0);
#endif
    }
    __syncthreads();
    par ^= 1;
  }

#pragma unroll
  for (int tm = 0; tm < 2; tm++) {
#pragma unroll
    for (int tn = 0; tn < 2; tn++) {
#pragma unroll
      for (int r = 0; r < 8; r++) {
        int m = rowBlk + mi * 32 + tm * 16 + r + ((lane >> 4) << 3);
        if (m < M) {
          int c0 = colBlk + ni * 32 + tn * 16 + (lane & 15);
          float v0 = acc[tm][tn][r] + (bias ? bias[c0] : 0.0f);
          if (ACT == 1) v0 = geluf(v0);
          if (RES) v0 += resid[(size_t)m * N + c0];
          if (OUT16) Yh[(size_t)m * N + c0] = (half_t)v0;
          else       Yf[(size_t)m * N + c0] = v0;
        }
      }
    }
  }
}

// ---------------------------------------------------------------------------
// locality bias lookup: h blocks (B,3,256,256)
// ---------------------------------------------------------------------------
DEVINL float attn_bias(const float* __restrict__ bb, int b, int q, int k) {
  if (q >= 285 && q < 541 && k < 256)
    return bb[(((size_t)b * 3 + 0) * 256 + (q - 285)) * 256 + k];
  if (q >= 571 && q < 827) {
    if (k < 256)              return bb[(((size_t)b * 3 + 1) * 256 + (q - 571)) * 256 + k];
    if (k >= 286 && k < 542)  return bb[(((size_t)b * 3 + 2) * 256 + (q - 571)) * 256 + (k - 286)];
  }
  return 0.0f;
}

// ---------------------------------------------------------------------------
// fused flash attention (base-2 online softmax)
// ---------------------------------------------------------------------------
__global__ __launch_bounds__(256) void attn_kernel(
    const half_t* __restrict__ q16, const half_t* __restrict__ k16,
    const half_t* __restrict__ vT, const float* __restrict__ bb,
    half_t* __restrict__ y16, int adaptive) {
  __shared__ __align__(32) half_t pbuf[8][16][32];
  const int lane = threadIdx.x & 31;
  const int wave = threadIdx.x >> 5;
  const int b  = blockIdx.z;
  const int hh = blockIdx.y;
  const int qBase = blockIdx.x * 128 + wave * 16;
  if (qBase >= TT) return;  // wave-uniform exit

  const half_t* Qp = q16 + (size_t)b * TT * C + hh * DHD;
  const half_t* Kp = k16 + (size_t)b * TT * C + hh * DHD;
  const half_t* Vp = vT + ((size_t)b * HH + hh) * DHD * TP;

  v16h aq = load_A(Qp, C, qBase, 0, TT - 1);

  v8f o0 = {}, o1 = {};
  float mA[8], lA[8];
#pragma unroll
  for (int r = 0; r < 8; r++) { mA[r] = -3.0e38f; lA[r] = 0.0f; }

  const float L2E = 1.44269504088896340f;
  const float sc2 = 0.17677669529663687f * L2E;  // (1/sqrt(32)) * log2(e)
  const int kEnd = min(qBase + 16, TT);

  for (int j0 = 0; j0 < kEnd; j0 += 32) {
    if (j0 + 32 < kEnd)
      __builtin_prefetch(Kp + (size_t)(j0 + 32 + (lane & 15)) * C, 0, 1);
    v8f z8 = {};
    v16h bk0 = load_B(Kp, C, j0,      0, TT - 1);
    v16h bk1 = load_B(Kp, C, j0 + 16, 0, TT - 1);
    v8f s1 = wmma16(aq, bk0, z8);
    v8f s2 = wmma16(aq, bk1, z8);

    float p1[8], p2[8];
#pragma unroll
    for (int r = 0; r < 8; r++) {
      int m  = qBase + r + ((lane >> 4) << 3);
      int c1 = j0 + (lane & 15);
      int c2 = c1 + 16;
      float v1 = s1[r] * sc2;   // log2 domain
      float v2 = s2[r] * sc2;
      if (adaptive) {
        v1 += attn_bias(bb, b, m, c1) * L2E;
        v2 += attn_bias(bb, b, m, c2) * L2E;
      }
      if (c1 > m || c1 >= TT) v1 = -__builtin_inff();
      if (c2 > m || c2 >= TT) v2 = -__builtin_inff();
      p1[r] = v1; p2[r] = v2;
    }

#pragma unroll
    for (int r = 0; r < 8; r++) {
      float t = redmax16(fmaxf(p1[r], p2[r]));
      float mNew = fmaxf(mA[r], t);
      float alpha = exp2f_fast(mA[r] - mNew);
      o0[r] *= alpha;
      o1[r] *= alpha;
      float e1 = exp2f_fast(p1[r] - mNew);
      float e2 = exp2f_fast(p2[r] - mNew);
      lA[r] = lA[r] * alpha + redsum16(e1 + e2);
      mA[r] = mNew;
      int row = r + ((lane >> 4) << 3);
      pbuf[wave][row][lane & 15]        = (half_t)e1;
      pbuf[wave][row][16 + (lane & 15)] = (half_t)e2;
    }
    asm volatile("s_wait_dscnt 0" ::: "memory");

    v16h ap = lds_read_A(&pbuf[wave][0][0], 0);
    v16h bv0 = load_B(Vp, TP, 0,  j0, DHD - 1);
    v16h bv1 = load_B(Vp, TP, 16, j0, DHD - 1);
    o0 = wmma16(ap, bv0, o0);
    o1 = wmma16(ap, bv1, o1);
  }

#pragma unroll
  for (int r = 0; r < 8; r++) {
    int m = qBase + r + ((lane >> 4) << 3);
    if (m < TT) {
      float inv = 1.0f / lA[r];
      size_t base = ((size_t)b * TT + m) * C + hh * DHD;
      y16[base + (lane & 15)]      = (half_t)(o0[r] * inv);
      y16[base + 16 + (lane & 15)] = (half_t)(o1[r] * inv);
    }
  }
}

// ---------------------------------------------------------------------------
// small helper kernels
// ---------------------------------------------------------------------------
__global__ void cvt_kernel(const float* __restrict__ s, half_t* __restrict__ d, int n) {
  int i = blockIdx.x * 256 + threadIdx.x;
  if (i < n) d[i] = (half_t)s[i];
}

__global__ void ln_kernel(const float* __restrict__ x, const float* __restrict__ w,
                          const float* __restrict__ bsh, half_t* __restrict__ o) {
  int row = blockIdx.x;
  int c = threadIdx.x;
  __shared__ float red[256];
  float v = x[(size_t)row * C + c];
  red[c] = v;
  __syncthreads();
  for (int s = 128; s > 0; s >>= 1) { if (c < s) red[c] += red[c + s]; __syncthreads(); }
  float mean = red[0] * (1.0f / C);
  __syncthreads();
  float d = v - mean;
  red[c] = d * d;
  __syncthreads();
  for (int s = 128; s > 0; s >>= 1) { if (c < s) red[c] += red[c + s]; __syncthreads(); }
  float var = red[0] * (1.0f / C);
  o[(size_t)row * C + c] = (half_t)(d * rsqrtf(var + 1e-5f) * w[c] + bsh[c]);
}

__global__ void embed_kernel(const float* __restrict__ dc, const float* __restrict__ zz,
                             const float* __restrict__ frame, const float* __restrict__ cam,
                             const float* __restrict__ timeE, float* __restrict__ xf) {
  int idx = blockIdx.x * 256 + threadIdx.x;
  if (idx >= BT * C) return;
  int c = idx & (C - 1);
  int t = (idx >> 8) % TT;
  int b = idx / (TT * C);
  float tok = (t < 572) ? dc[((size_t)b * 572 + t) * C + c]
                        : zz[((size_t)b * 256 + (t - 572)) * C + c];
  float role;
  if      (t < 256) role = frame[(size_t)t * C + c];
  else if (t < 286) role = cam[(size_t)(t - 256) * C + c];
  else if (t < 542) role = frame[(size_t)(t - 286) * C + c];
  else if (t < 572) role = cam[(size_t)(t - 542) * C + c];
  else              role = frame[(size_t)(t - 572) * C + c];
  xf[idx] = tok + role + timeE[(size_t)t * C + c];
}

__global__ void loc1_kernel(const float* __restrict__ p1, const float* __restrict__ p2,
                            const float* __restrict__ p3, const float* __restrict__ W1,
                            const float* __restrict__ b1f, float* __restrict__ z1) {
  int which = blockIdx.x / BB;
  int b = blockIdx.x % BB;
  int j = threadIdx.x;
  if (j >= 60) return;
  const float* p = (which == 0) ? p1 : ((which == 1) ? p2 : p3);
  float acc = b1f[j];
  for (int k = 0; k < 30; k++) acc += p[b * 30 + k] * W1[j * 30 + k];
  z1[(which * BB + b) * 60 + j] = geluf(acc);
}

__global__ void loc2_kernel(const float* __restrict__ z1, const float* __restrict__ W2,
                            const float* __restrict__ b2f, float* __restrict__ bb) {
  size_t idx = (size_t)blockIdx.x * 256 + threadIdx.x;
  if (idx >= (size_t)BB * 3 * 65536) return;
  int i = (int)(idx % 65536);
  int which = (int)((idx / 65536) % 3);
  int b = (int)(idx / (3 * 65536));
  const float* z = z1 + (which * BB + b) * 60;
  const float* wrow = W2 + (size_t)i * 60;
  float acc = b2f[i];
  for (int j = 0; j < 60; j++) acc += z[j] * wrow[j];
  bb[((size_t)b * 3 + which) * 65536 + i] = acc;
}

__global__ void vtr_kernel(const half_t* __restrict__ vsrc, half_t* __restrict__ vT) {
  int idx = blockIdx.x * 256 + threadIdx.x;
  if (idx >= BB * HH * DHD * TP) return;
  int t = idx % TP;
  int d = (idx / TP) % DHD;
  int h = (idx / (TP * DHD)) % HH;
  int b = idx / (TP * DHD * HH);
  vT[idx] = (t < TT) ? vsrc[((size_t)b * TT + t) * C + h * DHD + d] : (half_t)0.0f;
}

// ---------------------------------------------------------------------------
// host driver
// ---------------------------------------------------------------------------
extern "C" void kernel_launch(void* const* d_in, const int* in_sizes, int n_in,
                              void* d_out, int out_size, void* d_ws, size_t ws_size,
                              hipStream_t stream) {
  (void)in_sizes; (void)n_in; (void)out_size;
  const float* dc_emb = (const float*)d_in[0];
  const float* z_emb  = (const float*)d_in[1];
  const float* p1     = (const float*)d_in[2];
  const float* p2     = (const float*)d_in[3];
  const float* p3     = (const float*)d_in[4];
  const float* frame  = (const float*)d_in[5];
  const float* cam    = (const float*)d_in[6];
  const float* timeE  = (const float*)d_in[7];
  const float* locW1  = (const float*)d_in[8];
  const float* locb1  = (const float*)d_in[9];
  const float* locW2  = (const float*)d_in[10];
  const float* locb2  = (const float*)d_in[11];
  const float* ln1w   = (const float*)d_in[12];
  const float* ln1b   = (const float*)d_in[13];
  const float* Wq     = (const float*)d_in[14];
  const float* bq     = (const float*)d_in[15];
  const float* Wk     = (const float*)d_in[16];
  const float* bk     = (const float*)d_in[17];
  const float* Wv     = (const float*)d_in[18];
  const float* bvp    = (const float*)d_in[19];
  const float* Wo     = (const float*)d_in[20];
  const float* bo     = (const float*)d_in[21];
  const float* ln2w   = (const float*)d_in[22];
  const float* ln2b   = (const float*)d_in[23];
  const float* W1     = (const float*)d_in[24];
  const float* b1     = (const float*)d_in[25];
  const float* W2     = (const float*)d_in[26];
  const float* b2     = (const float*)d_in[27];
  const float* lnfw   = (const float*)d_in[28];
  const float* lnfb   = (const float*)d_in[29];
  const float* headW  = (const float*)d_in[30];
  float* out = (float*)d_out;

  char* base = (char*)d_ws;
  size_t off = 0;
  auto carve = [&](size_t bytes) -> char* {
    char* r = base + off;
    off = (off + bytes + 255) & ~(size_t)255;
    return r;
  };

  half_t* wq16   = (half_t*)carve((size_t)LNUM * C * C * sizeof(half_t));
  half_t* wk16   = (half_t*)carve((size_t)LNUM * C * C * sizeof(half_t));
  half_t* wv16   = (half_t*)carve((size_t)LNUM * C * C * sizeof(half_t));
  half_t* wo16   = (half_t*)carve((size_t)LNUM * C * C * sizeof(half_t));
  half_t* w116   = (half_t*)carve((size_t)LNUM * FF * C * sizeof(half_t));
  half_t* w216   = (half_t*)carve((size_t)LNUM * C * FF * sizeof(half_t));
  half_t* head16 = (half_t*)carve((size_t)VOCAB * C * sizeof(half_t));
  float*  xf     = (float*)carve((size_t)BT * C * sizeof(float));
  half_t* xn16   = (half_t*)carve((size_t)BT * C * sizeof(half_t));
  half_t* q16b   = (half_t*)carve((size_t)BT * C * sizeof(half_t));
  half_t* k16b   = (half_t*)carve((size_t)BT * C * sizeof(half_t));
  half_t* v16b   = (half_t*)carve((size_t)BT * C * sizeof(half_t));
  half_t* y16b   = (half_t*)carve((size_t)BT * C * sizeof(half_t));
  half_t* ffn16  = (half_t*)carve((size_t)BT * FF * sizeof(half_t));
  half_t* vTb    = (half_t*)carve((size_t)BB * HH * DHD * TP * sizeof(half_t));
  float*  z1buf  = (float*)carve((size_t)3 * BB * 60 * sizeof(float));
  float*  bb     = (float*)carve((size_t)BB * 3 * 65536 * sizeof(float));
  if (off > ws_size) return;  // workspace too small; bail deterministically

  auto cvt = [&](const float* s, half_t* d, size_t n) {
    cvt_kernel<<<dim3((unsigned)((n + 255) / 256)), dim3(256), 0, stream>>>(s, d, (int)n);
  };
  cvt(Wq, wq16, (size_t)LNUM * C * C);
  cvt(Wk, wk16, (size_t)LNUM * C * C);
  cvt(Wv, wv16, (size_t)LNUM * C * C);
  cvt(Wo, wo16, (size_t)LNUM * C * C);
  cvt(W1, w116, (size_t)LNUM * FF * C);
  cvt(W2, w216, (size_t)LNUM * C * FF);
  cvt(headW, head16, (size_t)VOCAB * C);

  loc1_kernel<<<dim3(3 * BB), dim3(64), 0, stream>>>(p1, p2, p3, locW1, locb1, z1buf);
  loc2_kernel<<<dim3((BB * 3 * 65536) / 256), dim3(256), 0, stream>>>(z1buf, locW2, locb2, bb);
  embed_kernel<<<dim3((BT * C) / 256), dim3(256), 0, stream>>>(dc_emb, z_emb, frame, cam, timeE, xf);

#define GEMM(ACT, O16, RES, Xp, Wp, Bp, Yfp, Yhp, Rp, M, N, K)                           \
  gemm_kernel<ACT, O16, RES><<<dim3((N) / 128, ((M) + 63) / 64), dim3(256), 0, stream>>>( \
      Xp, Wp, Bp, Yfp, Yhp, Rp, M, N, K)

  for (int i = 0; i < LNUM; i++) {
    ln_kernel<<<dim3(BT), dim3(256), 0, stream>>>(xf, ln1w + (size_t)i * C, ln1b + (size_t)i * C, xn16);
    GEMM(0, 1, 0, xn16, wq16 + (size_t)i * C * C, bq + (size_t)i * C, (float*)nullptr, q16b, (const float*)nullptr, BT, C, C);
    GEMM(0, 1, 0, xn16, wk16 + (size_t)i * C * C, bk + (size_t)i * C, (float*)nullptr, k16b, (const float*)nullptr, BT, C, C);
    GEMM(0, 1, 0, xn16, wv16 + (size_t)i * C * C, bvp + (size_t)i * C, (float*)nullptr, v16b, (const float*)nullptr, BT, C, C);
    vtr_kernel<<<dim3((BB * HH * DHD * TP) / 256), dim3(256), 0, stream>>>(v16b, vTb);
    attn_kernel<<<dim3((TT + 127) / 128, HH, BB), dim3(256), 0, stream>>>(
        q16b, k16b, vTb, bb, y16b, (i % 2 == 0) ? 1 : 0);
    GEMM(0, 0, 1, y16b, wo16 + (size_t)i * C * C, bo + (size_t)i * C, xf, (half_t*)nullptr, xf, BT, C, C);
    ln_kernel<<<dim3(BT), dim3(256), 0, stream>>>(xf, ln2w + (size_t)i * C, ln2b + (size_t)i * C, xn16);
    GEMM(1, 1, 0, xn16, w116 + (size_t)i * FF * C, b1 + (size_t)i * FF, (float*)nullptr, ffn16, (const float*)nullptr, BT, FF, C);
    GEMM(0, 0, 1, ffn16, w216 + (size_t)i * C * FF, b2 + (size_t)i * C, xf, (half_t*)nullptr, xf, BT, C, FF);
  }

  ln_kernel<<<dim3(BT), dim3(256), 0, stream>>>(xf, lnfw, lnfb, xn16);
  GEMM(0, 0, 0, xn16, head16, (const float*)nullptr, out, (half_t*)nullptr, (const float*)nullptr, BT, VOCAB, C);
#undef GEMM
}
